// MixAttention_84756884619938
// MI455X (gfx1250) — matrix operand order repaired and checked by reading, not compiled, not run on hardware
//
#include <hip/hip_runtime.h>
#include <hip/hip_bf16.h>
#include <math.h>

typedef __attribute__((ext_vector_type(16))) _Float16 v16h;
typedef __attribute__((ext_vector_type(8)))  _Float16 v8h;
typedef __attribute__((ext_vector_type(8)))  float    v8f;

#define C_    112
#define KP_   128
#define NH_   4
#define DH_   28
#define WS_   16
#define NW_   8
#define RES_  128
#define BS_   4
#define P2_   64
#define HW_   256
#define BW_   256
#define NROW_ 65536

__device__ __forceinline__ float gelu_f(float x){ return 0.5f*x*(1.0f+erff(x*0.70710678118f)); }
__device__ __forceinline__ float sigm_f(float x){ return 1.0f/(1.0f+__expf(-x)); }
__device__ __forceinline__ int winrow(int b,int Y,int X){
  return ((b*P2_ + (Y>>4)*NW_ + (X>>4))<<8) + ((Y&15)<<4) + (X&15);
}

// ---- WMMA operand loaders (layouts per CDNA5 ISA §7.12.2) ----
// A fragment: per lane m = lane&15, khi = 8*(lane>=16); the 16 halfs are the
// two contiguous runs [khi, khi+8) and [khi+16, khi+24) -> two b128 loads.
__device__ __forceinline__ v16h load_a16(const _Float16* __restrict__ A, int lda, int lane){
  int m = lane & 15; int khi = (lane >> 4) << 3;
  const _Float16* p = A + m*lda + khi;
  v8h lo = *(const v8h*)(p);
  v8h hi = *(const v8h*)(p + 16);
  v16h a;
  #pragma unroll
  for (int j=0;j<8;++j){ a[j] = lo[j]; a[8+j] = hi[j]; }
  return a;
}
// B fragment: per lane n = lane&15, khi = 8*(lane>=16); the 16 halfs are K rows
// [khi, khi+8) and [khi+16, khi+24) of column n (strided gather).
__device__ __forceinline__ v16h load_b16(const _Float16* __restrict__ B, int ldb, int lane){
  int n = lane & 15; int khi = (lane >> 4) << 3;
  const _Float16* p = B + khi*ldb + n;
  v16h b;
  #pragma unroll
  for (int j=0;j<8;++j){ b[j] = p[j*ldb]; b[8+j] = p[(16+j)*ldb]; }
  return b;
}

// =====================================================================
// Generic f16 GEMM:  C[M x 128]h * B[128 x N]h (+bias) -> f32.
// 8 waves / block, one 16x16 tile per wave, K fully unrolled (4 WMMAs).
// MAP=1 permutes output rows window-layout -> image-layout.
// =====================================================================
template<int MAP>
__global__ void gemm_f16k(const _Float16* __restrict__ A, const _Float16* __restrict__ B,
                          const float* __restrict__ bias, float* __restrict__ Cmat,
                          int N){
  int lane = threadIdx.x & 31, w = threadIdx.x >> 5;
  int m0 = blockIdx.x*128 + w*16;
  int n0 = blockIdx.y*16;
  v8f acc = {};
  #pragma unroll
  for (int k0 = 0; k0 < KP_; k0 += 32){
    v16h a = load_a16(A + (size_t)m0*KP_ + k0, KP_, lane);
    v16h b = load_b16(B + (size_t)k0*N + n0, N, lane);
    acc = __builtin_amdgcn_wmma_f32_16x16x32_f16(false,a,false,b,(short)0,acc,false,false);
  }
  int n = n0 + (lane&15);
  float bv = bias ? bias[n] : 0.0f;
  #pragma unroll
  for (int r=0;r<8;++r){
    int m = m0 + r + ((lane>>4)<<3);
    int row = m;
    if (MAP==1){
      int b_ = m >> 14, rem = m & 16383, p = rem >> 8, t = rem & 255;
      int Y = ((p>>3)<<4) + (t>>4), X = ((p&7)<<4) + (t&15);
      row = (b_*RES_ + Y)*RES_ + X;
    }
    Cmat[(size_t)row*N + n] = acc[r] + bv;
  }
}

// ---- staging: fp32 (rows x 112) -> f16 (rows x 128, zero pad), v8h stores ----
__global__ void stage_rows(const float* __restrict__ src, _Float16* __restrict__ dst, long total){
  long i = (long)blockIdx.x*blockDim.x + threadIdx.x;   // one v8h group per thread
  if (i >= total) return;
  long r = i >> 4; int g = (int)(i & 15);
  v8h o;
  if (g < 14){
    const float* s = src + r*C_ + g*8;
    #pragma unroll
    for (int j=0;j<8;++j) o[j] = (_Float16)s[j];
  } else {
    #pragma unroll
    for (int j=0;j<8;++j) o[j] = (_Float16)0.0f;
  }
  *(v8h*)(dst + r*KP_ + g*8) = o;
}

// ---- weight staging: fp32 (112 x N) -> f16 (128 x N, zero padded) ----
__global__ void stage_w(const float* __restrict__ src, _Float16* __restrict__ dst, int N){
  int i = blockIdx.x*blockDim.x + threadIdx.x;
  if (i >= KP_*N) return;
  int k = i / N;
  dst[i] = (k < C_) ? (_Float16)src[i] : (_Float16)0.0f;
}

// ---- LayerNorm over last dim (112), in place ----
__global__ void ln_rows(float* __restrict__ y, const float* __restrict__ g,
                        const float* __restrict__ b, int nrows){
  int r = blockIdx.x*blockDim.x + threadIdx.x;
  if (r >= nrows) return;
  float* p = y + (size_t)r*C_;
  float m = 0.f;
  for (int i=0;i<C_;++i) m += p[i];
  m *= (1.0f/C_);
  float v = 0.f;
  for (int i=0;i<C_;++i){ float d = p[i]-m; v += d*d; }
  v *= (1.0f/C_);
  float inv = rsqrtf(v + 1e-5f);
  for (int i=0;i<C_;++i) p[i] = (p[i]-m)*inv*g[i] + b[i];
}

// =====================================================================
// XCiT channel attention, one block per (window, head). All data in LDS.
// qkv channel layout for this branch: c = d*NH + h.
// =====================================================================
__global__ void xcit_k(const float* __restrict__ qkv, const float* __restrict__ sigma,
                       float* __restrict__ xc){
  extern __shared__ char smem[];
  float* sq  = (float*)smem;          // 256*28
  float* sk  = sq + HW_*DH_;
  float* sv  = sk + HW_*DH_;
  float* sA  = sv + HW_*DH_;          // 28*28
  float* srq = sA + DH_*DH_;          // 28
  float* srk = srq + DH_;             // 28
  int win = blockIdx.x, h = blockIdx.y, tid = threadIdx.x;
  {
    const float* base = qkv + (size_t)(win*HW_ + tid)*336;
    for (int d=0; d<DH_; ++d){
      sq[tid*DH_+d] = base[d*NH_+h];
      sk[tid*DH_+d] = base[112 + d*NH_+h];
      sv[tid*DH_+d] = base[224 + d*NH_+h];
    }
  }
  __syncthreads();
  if (tid < 2*DH_){
    int d = tid % DH_;
    const float* s = (tid < DH_) ? sq : sk;
    float acc = 0.f;
    for (int t=0;t<HW_;++t){ float v = s[t*DH_+d]; acc += v*v; }
    float nrm = fmaxf(sqrtf(acc), 1e-12f);
    if (tid < DH_) srq[d] = 1.0f/nrm; else srk[d] = 1.0f/nrm;
  }
  __syncthreads();
  float sg = sigma[(win & 63)*NH_ + h];
  for (int idx = tid; idx < DH_*DH_; idx += blockDim.x){
    int d = idx / DH_, e = idx % DH_;
    float acc = 0.f;
    for (int t=0;t<HW_;++t) acc += sk[t*DH_+d]*sq[t*DH_+e];
    sA[idx] = acc * srk[d] * srq[e] * sg;
  }
  __syncthreads();
  if (tid < DH_){                      // softmax over d (axis -2)
    int e = tid; float mx = -1e30f;
    for (int d=0;d<DH_;++d) mx = fmaxf(mx, sA[d*DH_+e]);
    float sm = 0.f;
    for (int d=0;d<DH_;++d){ float ev = __expf(sA[d*DH_+e]-mx); sA[d*DH_+e] = ev; sm += ev; }
    float inv = 1.0f/sm;
    for (int d=0;d<DH_;++d) sA[d*DH_+e] *= inv;
  }
  __syncthreads();
  {
    float* orow = xc + (size_t)(win*HW_ + tid)*C_;
    for (int e=0;e<DH_;++e){
      float acc = 0.f;
      for (int d=0;d<DH_;++d) acc += sv[tid*DH_+d]*sA[d*DH_+e];
      orow[e*NH_+h] = acc;            // xcit output channel = e*NH + h
    }
  }
}

// =====================================================================
// Spatial window attention, one block per (window, head), 8 waves.
// LDS: Q (256x32 f16) + K^T (32x256 f16) + V (256x32 f16) = 48KB,
//      64x256 f32 score panel (64KB) + f16 P panel (32KB) = 144KB total.
// A operands are contiguous b128 LDS loads; B operands are strided LDS gathers.
// qkv channel layout for this branch: c = h*DH + d.
// =====================================================================
__global__ void attn_k(const float* __restrict__ qkv, const float* __restrict__ pe,
                       float* __restrict__ outp){
  extern __shared__ char smem[];
  _Float16* sQ  = (_Float16*)smem;                                 // [256][32]
  _Float16* sKt = sQ + HW_*32;                                     // [32][256] (transposed)
  _Float16* sV  = sKt + 32*HW_;                                    // [256][32]
  float*    sS  = (float*)(smem + (size_t)3*HW_*32*sizeof(_Float16));   // [64][256]
  _Float16* sP  = (_Float16*)((char*)sS + (size_t)64*HW_*sizeof(float)); // [64][256]
  int win = blockIdx.x, h = blockIdx.y;
  int tid = threadIdx.x, lane = tid & 31, w = tid >> 5;
  const float qs = 0.18898223650461363f;   // 28^-0.5 folded into Q
  {
    const float* base = qkv + (size_t)(win*HW_ + tid)*336;
    _Float16 qrow[32], vrow[32];
    #pragma unroll
    for (int d=0; d<32; ++d){
      qrow[d] = (d<DH_) ? (_Float16)(base[h*DH_+d]*qs) : (_Float16)0.0f;
      vrow[d] = (d<DH_) ? (_Float16)(base[224 + h*DH_+d]) : (_Float16)0.0f;
      sKt[d*HW_ + tid] = (d<DH_) ? (_Float16)(base[112 + h*DH_+d]) : (_Float16)0.0f;
    }
    #pragma unroll
    for (int gq=0; gq<4; ++gq){
      *(v8h*)(sQ + tid*32 + gq*8) = *(v8h*)(qrow + gq*8);
      *(v8h*)(sV + tid*32 + gq*8) = *(v8h*)(vrow + gq*8);
    }
  }
  __syncthreads();
  const float* peh = pe + (size_t)h*HW_*HW_;
  for (int p=0; p<4; ++p){
    int i0 = p*64;
    // ---- S panel = Q[i0:i0+64] K^T (+pos_emb): 64 tiles over 8 waves ----
    for (int tt=0; tt<8; ++tt){
      int tile = w + tt*8;
      int mi = tile >> 4, nj = tile & 15;
      int ml0 = mi*16;
      v16h a = load_a16(sQ + (size_t)(i0+ml0)*32, 32, lane);
      v16h b = load_b16(sKt + nj*16, HW_, lane);   // (d=k, t=nj*16+(lane&15))
      v8f acc = {};
      acc = __builtin_amdgcn_wmma_f32_16x16x32_f16(false,a,false,b,(short)0,acc,false,false);
      int n = nj*16 + (lane&15);
      #pragma unroll
      for (int r=0;r<8;++r){
        int ml = ml0 + r + ((lane>>4)<<3);
        sS[ml*HW_ + n] = acc[r] + peh[(size_t)(i0+ml)*HW_ + n];
      }
    }
    __syncthreads();
    // ---- row softmax (wave32 shuffle reductions), write f16 P ----
    for (int rr=0; rr<8; ++rr){
      int ml = w*8 + rr;
      float* srow = sS + ml*HW_;
      float vbuf[8]; float mx = -1e30f;
      #pragma unroll
      for (int s=0;s<8;++s){ vbuf[s] = srow[lane + s*32]; mx = fmaxf(mx, vbuf[s]); }
      #pragma unroll
      for (int o=16;o>0;o>>=1) mx = fmaxf(mx, __shfl_xor(mx, o, 32));
      float sm = 0.f;
      #pragma unroll
      for (int s=0;s<8;++s){ vbuf[s] = __expf(vbuf[s]-mx); sm += vbuf[s]; }
      #pragma unroll
      for (int o=16;o>0;o>>=1) sm += __shfl_xor(sm, o, 32);
      float inv = 1.0f/sm;
      _Float16* prow = sP + ml*HW_;
      #pragma unroll
      for (int s=0;s<8;++s) prow[lane + s*32] = (_Float16)(vbuf[s]*inv);
    }
    __syncthreads();
    // ---- O panel = P (64x256) @ V (256x32), one tile per wave ----
    {
      int mi = w >> 1, nj = w & 1;
      v8f acc = {};
      #pragma unroll
      for (int ks=0; ks<8; ++ks){
        v16h a = load_a16(sP + (size_t)(mi*16)*HW_ + ks*32, HW_, lane);
        v16h b = load_b16(sV + (size_t)(ks*32)*32 + nj*16, 32, lane);
        acc = __builtin_amdgcn_wmma_f32_16x16x32_f16(false,a,false,b,(short)0,acc,false,false);
      }
      int d = nj*16 + (lane&15);
      if (d < DH_){
        #pragma unroll
        for (int r=0;r<8;++r){
          int ml = mi*16 + r + ((lane>>4)<<3);
          outp[(size_t)(win*HW_ + i0 + ml)*C_ + h*DH_ + d] = acc[r];
        }
      }
    }
    __syncthreads();
  }
}

// ---- depthwise 3x3 on v-image (v read from qkv via window mapping) ----
__global__ void conv_vimg(const float* __restrict__ qkv, const float* __restrict__ wpe,
                          float* __restrict__ t1){
  int i = blockIdx.x*blockDim.x + threadIdx.x;
  if (i >= NROW_*C_) return;
  int c = i % C_, pix = i / C_;
  int b = pix >> 14, Y = (pix >> 7) & 127, X = pix & 127;
  float acc = 0.f;
  for (int ky=0;ky<3;++ky){
    int Yn = Y + ky - 1; if ((unsigned)Yn >= RES_) continue;
    for (int kx=0;kx<3;++kx){
      int Xn = X + kx - 1; if ((unsigned)Xn >= RES_) continue;
      acc += qkv[(size_t)winrow(b,Yn,Xn)*336 + 224 + c] * wpe[c*9 + ky*3 + kx];
    }
  }
  t1[i] = acc;
}

// ---- x_cnn_img += dw3(gelu(t1), pe2) ----
__global__ void conv_pe2_add(const float* __restrict__ t1, const float* __restrict__ wpe,
                             float* __restrict__ ximg){
  int i = blockIdx.x*blockDim.x + threadIdx.x;
  if (i >= NROW_*C_) return;
  int c = i % C_, pix = i / C_;
  int b = pix >> 14, Y = (pix >> 7) & 127, X = pix & 127;
  float acc = 0.f;
  for (int ky=0;ky<3;++ky){
    int Yn = Y + ky - 1; if ((unsigned)Yn >= RES_) continue;
    for (int kx=0;kx<3;++kx){
      int Xn = X + kx - 1; if ((unsigned)Xn >= RES_) continue;
      acc += gelu_f(t1[(size_t)((b*RES_+Yn)*RES_+Xn)*C_ + c]) * wpe[c*9 + ky*3 + kx];
    }
  }
  ximg[i] += acc;
}

// ---- dst = dw3(ximg, dw_w) + dw_b ----
__global__ void conv_dw_b(const float* __restrict__ src, const float* __restrict__ wdw,
                          const float* __restrict__ bdw, float* __restrict__ dst){
  int i = blockIdx.x*blockDim.x + threadIdx.x;
  if (i >= NROW_*C_) return;
  int c = i % C_, pix = i / C_;
  int b = pix >> 14, Y = (pix >> 7) & 127, X = pix & 127;
  float acc = 0.f;
  for (int ky=0;ky<3;++ky){
    int Yn = Y + ky - 1; if ((unsigned)Yn >= RES_) continue;
    for (int kx=0;kx<3;++kx){
      int Xn = X + kx - 1; if ((unsigned)Xn >= RES_) continue;
      acc += src[(size_t)((b*RES_+Yn)*RES_+Xn)*C_ + c] * wdw[c*9 + ky*3 + kx];
    }
  }
  dst[i] = acc + bdw[c];
}

// ---- gap over pixels then channel-interaction MLP -> ci logits ----
__global__ void gap_ci_k(const float* __restrict__ ximg, const float* __restrict__ w1,
                         const float* __restrict__ b1, const float* __restrict__ w2,
                         const float* __restrict__ b2, float* __restrict__ ci){
  __shared__ float gap[C_];
  __shared__ float c1[28];
  int b = blockIdx.x, tid = threadIdx.x;
  if (tid < C_){
    float acc = 0.f;
    const float* base = ximg + (size_t)b*16384*C_ + tid;
    for (int pj=0; pj<16384; ++pj) acc += base[(size_t)pj*C_];
    gap[tid] = acc * (1.0f/16384.0f);
  }
  __syncthreads();
  if (tid < 28){
    float acc = b1[tid];
    for (int c=0;c<C_;++c) acc += gap[c]*w1[tid*C_+c];
    c1[tid] = gelu_f(acc);
  }
  __syncthreads();
  if (tid < C_){
    float acc = b2[tid];
    for (int j=0;j<28;++j) acc += c1[j]*w2[tid*28+j];
    ci[b*C_+tid] = acc;
  }
}

// ---- spatial-interaction gate: per-pixel 1x1 MLP on attention output ----
__global__ void si_k(const float* __restrict__ outw, const float* __restrict__ w1,
                     const float* __restrict__ b1, const float* __restrict__ w2,
                     const float* __restrict__ b2, float* __restrict__ si){
  int pix = blockIdx.x*blockDim.x + threadIdx.x;
  if (pix >= BS_*16384) return;
  int b = pix >> 14, Y = (pix >> 7) & 127, X = pix & 127;
  const float* row = outw + (size_t)winrow(b,Y,X)*C_;
  float acc = b2[0];
  for (int j=0;j<14;++j){
    float s = b1[j];
    for (int c=0;c<C_;++c) s += row[c]*w1[j*C_+c];
    acc += w2[j]*gelu_f(s);
  }
  si[pix] = acc;
}

__global__ void gate_k(float* __restrict__ ximg, const float* __restrict__ si){
  int i = blockIdx.x*blockDim.x + threadIdx.x;
  if (i >= NROW_*C_) return;
  ximg[i] *= sigm_f(si[i / C_]);
}

// ---- proj1 (C->56) per pixel, write f16 into concat channels [56..112) ----
__global__ void proj1_k(const float* __restrict__ xin, const float* __restrict__ w,
                        const float* __restrict__ bb, _Float16* __restrict__ concat){
  int i = blockIdx.x*blockDim.x + threadIdx.x;
  if (i >= NROW_*56) return;
  int j = i % 56, pix = i / 56;
  int b = pix >> 14, Y = (pix >> 7) & 127, X = pix & 127;
  const float* row = xin + (size_t)pix*C_;
  float acc = bb[j];
  for (int c=0;c<C_;++c) acc += row[c]*w[j*C_+c];
  _Float16* crow = concat + (size_t)winrow(b,Y,X)*KP_;
  crow[56 + j] = (_Float16)acc;
  if (j < 16) crow[112 + j] = (_Float16)0.0f;   // zero K-padding
}

// ---- ci-gated proj2 (C->56), write f16 into concat channels [0..56) ----
__global__ void proj2_k(const float* __restrict__ outw, const float* __restrict__ ci,
                        const float* __restrict__ w, const float* __restrict__ bb,
                        _Float16* __restrict__ concat){
  int i = blockIdx.x*blockDim.x + threadIdx.x;
  if (i >= NROW_*56) return;
  int j = i % 56; int row = i / 56;
  int b = row >> 14;
  const float* r = outw + (size_t)row*C_;
  const float* cib = ci + b*C_;
  float acc = bb[j];
  for (int c=0;c<C_;++c) acc += sigm_f(cib[c])*r[c]*w[j*C_+c];
  concat[(size_t)row*KP_ + j] = (_Float16)acc;
}

extern "C" void kernel_launch(void* const* d_in, const int* in_sizes, int n_in,
                              void* d_out, int out_size, void* d_ws, size_t ws_size,
                              hipStream_t stream) {
  const float* x           = (const float*)d_in[0];
  const float* pos_emb     = (const float*)d_in[1];
  const float* proj_cnn_w  = (const float*)d_in[2];
  const float* proj_cnn_b  = (const float*)d_in[3];
  const float* ln_cnn_g    = (const float*)d_in[4];
  const float* ln_cnn_b    = (const float*)d_in[5];
  const float* qkv_cnn_w   = (const float*)d_in[6];
  const float* sigma_cnn   = (const float*)d_in[7];
  const float* out_cnn_w   = (const float*)d_in[8];
  const float* out_cnn_b   = (const float*)d_in[9];
  const float* pe1_w       = (const float*)d_in[10];
  const float* pe2_w       = (const float*)d_in[11];
  const float* dw_w        = (const float*)d_in[12];
  const float* dw_b        = (const float*)d_in[13];
  const float* proj1_w     = (const float*)d_in[14];
  const float* proj1_b     = (const float*)d_in[15];
  const float* proj_attn_w = (const float*)d_in[16];
  const float* proj_attn_b = (const float*)d_in[17];
  const float* ln_attn_g   = (const float*)d_in[18];
  const float* ln_attn_b   = (const float*)d_in[19];
  const float* qkv_w       = (const float*)d_in[20];
  const float* out_w       = (const float*)d_in[21];
  const float* out_b       = (const float*)d_in[22];
  const float* proj2_w     = (const float*)d_in[23];
  const float* proj2_b     = (const float*)d_in[24];
  const float* ci1_w       = (const float*)d_in[25];
  const float* ci1_b       = (const float*)d_in[26];
  const float* ci2_w       = (const float*)d_in[27];
  const float* ci2_b       = (const float*)d_in[28];
  const float* si1_w       = (const float*)d_in[29];
  const float* si1_b       = (const float*)d_in[30];
  const float* si2_w       = (const float*)d_in[31];
  const float* si2_b       = (const float*)d_in[32];
  const float* proj_w      = (const float*)d_in[33];
  const float* proj_b      = (const float*)d_in[34];
  (void)in_sizes; (void)n_in; (void)out_size; (void)ws_size;
  float* out = (float*)d_out;
  char* ws = (char*)d_ws;

  size_t off = 0;
  auto alloc = [&](size_t bytes)->char*{
    char* p = ws + off;
    off = (off + bytes + 255) & ~(size_t)255;
    return p;
  };
  _Float16* xh  = (_Float16*)alloc((size_t)NROW_*KP_*2);
  _Float16* wpc = (_Float16*)alloc((size_t)KP_*112*2);
  _Float16* wqc = (_Float16*)alloc((size_t)KP_*336*2);
  _Float16* woc = (_Float16*)alloc((size_t)KP_*112*2);
  _Float16* wpa = (_Float16*)alloc((size_t)KP_*112*2);
  _Float16* wqa = (_Float16*)alloc((size_t)KP_*336*2);
  _Float16* wow = (_Float16*)alloc((size_t)KP_*112*2);
  _Float16* wpj = (_Float16*)alloc((size_t)KP_*112*2);
  float* y1   = (float*)alloc((size_t)NROW_*C_*4);   // x_cnn -> xc -> attn_out
  float* y2   = (float*)alloc((size_t)NROW_*C_*4);   // x_at  -> xcnn2
  float* qkv  = (float*)alloc((size_t)NROW_*336*4);  // shared by both branches
  _Float16* stg = (_Float16*)alloc((size_t)NROW_*KP_*2); // reusable f16 staging
  float* ximg = (float*)alloc((size_t)NROW_*C_*4);   // x_cnn image (NHWC)
  float* t1   = (float*)alloc((size_t)NROW_*C_*4);   // pe tmp -> out_win
  float* si   = (float*)alloc((size_t)NROW_*4);
  float* ci   = (float*)alloc((size_t)BS_*C_*4);

  const int TPB = 256;
  const long TOTV = (long)NROW_*16;   // v8h groups per staging pass
  const int  TOTC = NROW_*C_;
  const long T56  = (long)NROW_*56;
  dim3 g112(NROW_/128, 112/16), g336(NROW_/128, 336/16);
  size_t xcit_lds = (size_t)(3*HW_*DH_ + DH_*DH_ + 2*DH_)*sizeof(float);      // ~87KB
  size_t attn_lds = (size_t)3*HW_*32*2 + (size_t)64*HW_*4 + (size_t)64*HW_*2; // 144KB

  // stage input + weights to padded f16
  stage_rows<<<dim3((unsigned)((TOTV+TPB-1)/TPB)),TPB,0,stream>>>(x, xh, TOTV);
  stage_w<<<(KP_*112+TPB-1)/TPB,TPB,0,stream>>>(proj_cnn_w,  wpc, 112);
  stage_w<<<(KP_*336+TPB-1)/TPB,TPB,0,stream>>>(qkv_cnn_w,   wqc, 336);
  stage_w<<<(KP_*112+TPB-1)/TPB,TPB,0,stream>>>(out_cnn_w,   woc, 112);
  stage_w<<<(KP_*112+TPB-1)/TPB,TPB,0,stream>>>(proj_attn_w, wpa, 112);
  stage_w<<<(KP_*336+TPB-1)/TPB,TPB,0,stream>>>(qkv_w,       wqa, 336);
  stage_w<<<(KP_*112+TPB-1)/TPB,TPB,0,stream>>>(out_w,       wow, 112);
  stage_w<<<(KP_*112+TPB-1)/TPB,TPB,0,stream>>>(proj_w,      wpj, 112);

  // ---- channel (XCiT) branch ----
  gemm_f16k<0><<<g112,256,0,stream>>>(xh, wpc, proj_cnn_b, y1, 112);
  ln_rows<<<NROW_/TPB,TPB,0,stream>>>(y1, ln_cnn_g, ln_cnn_b, NROW_);
  stage_rows<<<dim3((unsigned)((TOTV+TPB-1)/TPB)),TPB,0,stream>>>(y1, stg, TOTV);
  gemm_f16k<0><<<g336,256,0,stream>>>(stg, wqc, nullptr, qkv, 336);
  xcit_k<<<dim3(BW_,NH_),256,xcit_lds,stream>>>(qkv, sigma_cnn, y1);
  stage_rows<<<dim3((unsigned)((TOTV+TPB-1)/TPB)),TPB,0,stream>>>(y1, stg, TOTV);
  gemm_f16k<1><<<g112,256,0,stream>>>(stg, woc, out_cnn_b, ximg, 112);  // window->image rows
  conv_vimg<<<(TOTC+TPB-1)/TPB,TPB,0,stream>>>(qkv, pe1_w, t1);
  conv_pe2_add<<<(TOTC+TPB-1)/TPB,TPB,0,stream>>>(t1, pe2_w, ximg);
  gap_ci_k<<<BS_,128,0,stream>>>(ximg, ci1_w, ci1_b, ci2_w, ci2_b, ci);

  // ---- spatial window-attention branch ----
  gemm_f16k<0><<<g112,256,0,stream>>>(xh, wpa, proj_attn_b, y2, 112);
  ln_rows<<<NROW_/TPB,TPB,0,stream>>>(y2, ln_attn_g, ln_attn_b, NROW_);
  stage_rows<<<dim3((unsigned)((TOTV+TPB-1)/TPB)),TPB,0,stream>>>(y2, stg, TOTV);
  gemm_f16k<0><<<g336,256,0,stream>>>(stg, wqa, nullptr, qkv, 336);
  attn_k<<<dim3(BW_,NH_),256,attn_lds,stream>>>(qkv, pos_emb, y1);
  stage_rows<<<dim3((unsigned)((TOTV+TPB-1)/TPB)),TPB,0,stream>>>(y1, stg, TOTV);
  gemm_f16k<0><<<g112,256,0,stream>>>(stg, wow, out_b, t1, 112);        // out_win
  si_k<<<(NROW_+TPB-1)/TPB,TPB,0,stream>>>(t1, si1_w, si1_b, si2_w, si2_b, si);

  // ---- bidirectional gating / fusion ----
  gate_k<<<(TOTC+TPB-1)/TPB,TPB,0,stream>>>(ximg, si);
  conv_dw_b<<<(TOTC+TPB-1)/TPB,TPB,0,stream>>>(ximg, dw_w, dw_b, y2);
  proj1_k<<<dim3((unsigned)((T56+TPB-1)/TPB)),TPB,0,stream>>>(y2, proj1_w, proj1_b, stg);
  proj2_k<<<dim3((unsigned)((T56+TPB-1)/TPB)),TPB,0,stream>>>(t1, ci, proj2_w, proj2_b, stg);
  gemm_f16k<0><<<g112,256,0,stream>>>(stg, wpj, proj_b, out, 112);
}
